// TextEncoder_78932908966625
// MI455X (gfx1250) — matrix operand order repaired and checked by reading, not compiled, not run on hardware
//
#include <hip/hip_runtime.h>

// ---------------- problem constants ----------------
#define CCH   512          // channels
#define KW    5            // conv kernel width
#define DEPTH 3
#define BB    32           // batch
#define TT    1024         // time
#define HH    256          // per-direction hidden
#define G4H   1024         // 4*H
#define PADW  2            // (K-1)/2
#define TPAD  (TT + 2*PADW)

typedef __bf16 bf16;
typedef __attribute__((ext_vector_type(16))) __bf16 v16bf;
typedef __attribute__((ext_vector_type(8)))  __bf16 v8bf;
typedef __attribute__((ext_vector_type(8)))  float  v8f;

// ---- gfx1250 async global->LDS path (guarded; falls back to manual copy) ----
#if defined(__has_builtin)
#  if __has_builtin(__builtin_amdgcn_global_load_async_to_lds_b64) && \
      __has_builtin(__builtin_amdgcn_global_load_async_to_lds_b128) && \
      __has_builtin(__builtin_amdgcn_s_wait_asynccnt)
#    define ASYNC_LDS 1
#  endif
#endif
#ifndef ASYNC_LDS
#define ASYNC_LDS 0
#endif

#if ASYNC_LDS
// prototypes (from hipcc diagnostics): (global v2i*/v4i* src, lds v2i*/v4i* dst, Ii, Ii)
typedef int vi2 __attribute__((vector_size(8)));
typedef int vi4 __attribute__((vector_size(16)));
typedef __attribute__((address_space(1))) vi2 gvi2;
typedef __attribute__((address_space(3))) vi2 lvi2;
typedef __attribute__((address_space(1))) vi4 gvi4;
typedef __attribute__((address_space(3))) vi4 lvi4;
#endif

__device__ __forceinline__ void stage8(const bf16* g, bf16* l) {   // 8 bytes
#if ASYNC_LDS
    __builtin_amdgcn_global_load_async_to_lds_b64((gvi2*)g, (lvi2*)l, 0, 0);
#else
    *(unsigned long long*)l = *(const unsigned long long*)g;
#endif
}
__device__ __forceinline__ void stage16(const bf16* g, bf16* l) {  // 16 bytes
#if ASYNC_LDS
    __builtin_amdgcn_global_load_async_to_lds_b128((gvi4*)g, (lvi4*)l, 0, 0);
#else
    *(float4*)l = *(const float4*)g;
#endif
}
__device__ __forceinline__ void stage_wait() {
#if ASYNC_LDS
    __builtin_amdgcn_s_wait_asynccnt(0);
#endif
}

__device__ __forceinline__ v16bf mk16(v8bf lo, v8bf hi) {
    v16bf r;
#pragma unroll
    for (int i = 0; i < 8; ++i) { r[i] = lo[i]; r[i + 8] = hi[i]; }
    return r;
}

__device__ __forceinline__ v8f wmma_bf16(v16bf a, v16bf b, v8f c) {
    return __builtin_amdgcn_wmma_f32_16x16x32_bf16(
        false, a, false, b, (short)0, c, false, false);
}

__device__ __forceinline__ float sigm(float x) { return 1.0f / (1.0f + __expf(-x)); }

// ---------------- f32 -> bf16 convert ----------------
__global__ void f2bf_kernel(const float* __restrict__ in, bf16* __restrict__ out, int n) {
    int i = blockIdx.x * 256 + threadIdx.x;
    if (i < n) out[i] = (bf16)in[i];
}

// ---------------- embedding gather + mask ----------------
__global__ void emb_kernel(const int* __restrict__ x, const unsigned char* __restrict__ m,
                           const float* __restrict__ emb, float* __restrict__ actF) {
    int bt  = blockIdx.x;                 // b*T + t
    int sym = x[bt];
    bool pad = m[bt] != 0;
    const float* row = emb + (size_t)sym * CCH;
#pragma unroll
    for (int q = 0; q < 2; ++q) {
        int c = threadIdx.x + q * 256;
        actF[(size_t)bt * CCH + c] = pad ? 0.0f : row[c];
    }
}

// ---------------- pack f32 act -> padded bf16 [B, T+4, C] ----------------
__global__ void pack_kernel(const float* __restrict__ actF, bf16* __restrict__ padB) {
    int idx = blockIdx.x * 256 + threadIdx.x;
    int total = BB * TPAD * CCH;
    if (idx >= total) return;
    int c = idx % CCH;
    int r = (idx / CCH) % TPAD;
    int b = idx / (CCH * TPAD);
    int t = r - PADW;
    float v = (t >= 0 && t < TT) ? actF[((size_t)b * TT + t) * CCH + c] : 0.0f;
    padB[idx] = (bf16)v;
}

// ---------------- conv1d as 5 shifted WMMA GEMMs ----------------
// grid: (B*T/32, C/128); block 256 = 8 waves; each wave: 2 M-tiles x 1 N-tile
// (two t-tiles share every B fragment -> halved L2 weight traffic per wmma)
__global__ __launch_bounds__(256)
void conv_wmma_kernel(const bf16* __restrict__ Apad,           // [B, T+4, C] bf16
                      const bf16* __restrict__ W,              // [K, C, C] bf16 (tap, ci, co)
                      const float* __restrict__ bias,          // [C]
                      float* __restrict__ Out) {               // [B, T, C] f32
    __shared__ bf16 sA[36 * 32];                               // 36 t-rows x 32 ci
    const int lane = threadIdx.x & 31;
    const int wave = threadIdx.x >> 5;
    const int mt2 = blockIdx.x;             // 0 .. B*T/32-1
    const int b   = mt2 >> 5;               // 32 double-tiles per batch row
    const int t0  = (mt2 & 31) << 5;
    const int co0 = blockIdx.y * 128 + wave * 16;
    const int rowL = lane & 15;
    const int kg   = (lane >> 4) * 8;

    v8f acc0 = {}, acc1 = {};
    for (int ks = 0; ks < 16; ++ks) {
        const int ci = ks * 32;
        {   // stage 36x32 bf16 tile: 288 8-byte chunks over 256 threads
            int r  = threadIdx.x >> 3;
            int ch = threadIdx.x & 7;
            stage8(Apad + ((size_t)(b * TPAD + t0 + r)) * CCH + ci + ch * 4,
                   &sA[r * 32 + ch * 4]);
            if (threadIdx.x < 32) {
                int r2 = 32 + (threadIdx.x >> 3);
                stage8(Apad + ((size_t)(b * TPAD + t0 + r2)) * CCH + ci + ch * 4,
                       &sA[r2 * 32 + ch * 4]);
            }
        }
        stage_wait();
        __syncthreads();
        if (ks < 15)                        // hint next B tile toward the caches
            __builtin_prefetch(W + ((size_t)(ci + 32) + lane) * CCH + co0, 0, 0);
#pragma unroll
        for (int k = 0; k < KW; ++k) {
            int r0 = k + rowL;
            v16bf af0 = mk16(*(const v8bf*)&sA[r0 * 32 + kg],
                             *(const v8bf*)&sA[r0 * 32 + kg + 16]);
            int r1 = 16 + k + rowL;
            v16bf af1 = mk16(*(const v8bf*)&sA[r1 * 32 + kg],
                             *(const v8bf*)&sA[r1 * 32 + kg + 16]);
            v16bf bf;
#pragma unroll
            for (int s = 0; s < 16; ++s)
                bf[s] = W[((size_t)k * CCH + ci + kg + s + (s & 8)) * CCH + co0 + rowL];
            acc0 = wmma_bf16(af0, bf, acc0);
            acc1 = wmma_bf16(af1, bf, acc1);
        }
        __syncthreads();
    }
    const int n = co0 + rowL;
    const float bv = bias[n];
    const int mbase = (lane >> 4) << 3;
#pragma unroll
    for (int v = 0; v < 8; ++v) {
        Out[((size_t)(b * TT + t0 + mbase + v)) * CCH + n]      = acc0[v] + bv;
        Out[((size_t)(b * TT + t0 + 16 + mbase + v)) * CCH + n] = acc1[v] + bv;
    }
}

// ---------------- LayerNorm over time + leaky-relu + mask ----------------
__global__ __launch_bounds__(256)
void ln_lrelu_kernel(const float* __restrict__ In, const float* __restrict__ scale,
                     const float* __restrict__ bias, const unsigned char* __restrict__ m,
                     float* __restrict__ Outp) {
    __shared__ float red[256];
    const int bc = blockIdx.x;
    const int b = bc >> 9, c = bc & 511;
    const int tid = threadIdx.x;
    float vals[4]; float s = 0.0f;
#pragma unroll
    for (int q = 0; q < 4; ++q) {
        int t = tid + q * 256;
        vals[q] = In[((size_t)b * TT + t) * CCH + c];
        s += vals[q];
    }
    red[tid] = s; __syncthreads();
    for (int off = 128; off; off >>= 1) { if (tid < off) red[tid] += red[tid + off]; __syncthreads(); }
    const float mu = red[0] * (1.0f / TT);
    __syncthreads();
    float s2 = 0.0f;
#pragma unroll
    for (int q = 0; q < 4; ++q) { float d = vals[q] - mu; s2 += d * d; }
    red[tid] = s2; __syncthreads();
    for (int off = 128; off; off >>= 1) { if (tid < off) red[tid] += red[tid + off]; __syncthreads(); }
    const float inv = rsqrtf(red[0] * (1.0f / TT) + 1e-5f);
#pragma unroll
    for (int q = 0; q < 4; ++q) {
        int t = tid + q * 256;
        float z = (vals[q] - mu) * inv * scale[t] + bias[t];
        z = (z > 0.0f) ? z : 0.2f * z;                 // leaky relu 0.2
        if (m[b * TT + t]) z = 0.0f;
        Outp[((size_t)b * TT + t) * CCH + c] = z;
    }
}

// ---------------- xg = h @ Wx + b  (bf16 output) ----------------
// grid: (B*T/32, 4H/128); block 256; each wave 2 M-tiles sharing B fragments
__global__ __launch_bounds__(256)
void xg_wmma_kernel(const bf16* __restrict__ Apad,             // [B, T+4, C] bf16 (rows +PAD)
                    const bf16* __restrict__ Wx,               // [C, 4H] bf16
                    const float* __restrict__ bias,            // [4H]
                    bf16* __restrict__ Xg) {                   // [B*T, 4H] bf16
    __shared__ bf16 sA[32 * 32];
    const int lane = threadIdx.x & 31;
    const int wave = threadIdx.x >> 5;
    const int mt2 = blockIdx.x;
    const int b   = mt2 >> 5;
    const int t0  = (mt2 & 31) << 5;
    const int co0 = blockIdx.y * 128 + wave * 16;
    const int rowL = lane & 15;
    const int kg   = (lane >> 4) * 8;

    v8f acc0 = {}, acc1 = {};
    for (int ks = 0; ks < 16; ++ks) {
        const int ci = ks * 32;
        {   // stage 32x32 tile: exactly 256 8-byte chunks
            int r  = threadIdx.x >> 3;
            int ch = threadIdx.x & 7;
            stage8(Apad + ((size_t)(b * TPAD + PADW + t0 + r)) * CCH + ci + ch * 4,
                   &sA[r * 32 + ch * 4]);
        }
        stage_wait();
        __syncthreads();
        v16bf af0 = mk16(*(const v8bf*)&sA[rowL * 32 + kg],
                         *(const v8bf*)&sA[rowL * 32 + kg + 16]);
        int r1 = 16 + rowL;
        v16bf af1 = mk16(*(const v8bf*)&sA[r1 * 32 + kg],
                         *(const v8bf*)&sA[r1 * 32 + kg + 16]);
        v16bf bf;
#pragma unroll
        for (int s = 0; s < 16; ++s)
            bf[s] = Wx[((size_t)(ci + kg + s + (s & 8))) * G4H + co0 + rowL];
        acc0 = wmma_bf16(af0, bf, acc0);
        acc1 = wmma_bf16(af1, bf, acc1);
        __syncthreads();
    }
    const int n = co0 + rowL;
    const float bv = bias[n];
    const int mbase = (lane >> 4) << 3;
#pragma unroll
    for (int v = 0; v < 8; ++v) {
        Xg[((size_t)(b * TT + t0 + mbase + v)) * G4H + n]      = (bf16)(acc0[v] + bv);
        Xg[((size_t)(b * TT + t0 + 16 + mbase + v)) * G4H + n] = (bf16)(acc1[v] + bv);
    }
}

// ---------------- persistent LSTM scan (one workgroup, 32 waves) ----------------
// Per step: issue async xg[t]->LDS, WMMA the recurrence on h_prev (LDS),
// s_wait_asynccnt + barrier, gates=acc+xg (all LDS), cell update, write d_out.
__global__ __launch_bounds__(1024)
void lstm_kernel(const bf16* __restrict__ Xg,                  // [B*T, 4H] bf16
                 const bf16* __restrict__ Wh,                  // [H, 4H] bf16
                 const unsigned char* __restrict__ m,
                 float* __restrict__ Out, int reverse, int chOff) {
    extern __shared__ char smem[];
    bf16*  hp     = (bf16*)smem;                               // 32*256 bf16   (16 KB)
    float* gates  = (float*)(smem + (size_t)BB * HH * 2);      // 32*1024 f32   (128 KB)
    bf16*  xstage = (bf16*)(smem + (size_t)BB * HH * 2
                                 + (size_t)BB * G4H * 4);      // 32*1024 bf16  (64 KB)

    const int tid = threadIdx.x;
    const int lane = tid & 31;
    const int wave = tid >> 5;
    const int mt = wave >> 4;                                  // 0/1: batch rows 0-15 / 16-31
    const int ntbase = (wave & 15) * 4;                        // 4 n-tiles per wave
    const int rowL = lane & 15;
    const int kg   = (lane >> 4) * 8;
    const int arow = (mt * 16 + rowL) * HH;
    const int brow = mt * 16 + ((lane >> 4) << 3);

    float creg[8];
#pragma unroll
    for (int q = 0; q < 8; ++q) creg[q] = 0.0f;
#pragma unroll
    for (int q = 0; q < 8; ++q) hp[tid * 8 + q] = (bf16)0.0f;
    __syncthreads();

    for (int s = 0; s < TT; ++s) {
        const int t = reverse ? (TT - 1 - s) : s;
        // ---- kick off async DMA of xg[:, t, :] into LDS (overlaps with WMMA) ----
#pragma unroll
        for (int q = 0; q < 4; ++q) {
            int e  = (tid * 4 + q) * 8;                        // element offset, 16B chunks
            int xb = e >> 10, xn = e & 1023;
            stage16(Xg + ((size_t)(xb * TT + t)) * G4H + xn, xstage + e);
        }
        // ---- recurrence GEMM: [32,256] x [256,1024] in bf16 WMMA ----
        v8f acc0 = {}, acc1 = {}, acc2 = {}, acc3 = {};
#pragma unroll
        for (int ks = 0; ks < 8; ++ks) {
            int klo = ks * 32 + kg;
            v16bf af = mk16(*(const v8bf*)&hp[arow + klo],
                            *(const v8bf*)&hp[arow + klo + 16]);
            const bf16* wr = Wh + ((size_t)(ks * 32 + kg)) * G4H + rowL;
            v16bf b0, b1, b2, b3;
#pragma unroll
            for (int s = 0; s < 16; ++s) {
                const size_t o = (size_t)(s + (s & 8)) * G4H;
                b0[s] = wr[o + (ntbase + 0) * 16];
                b1[s] = wr[o + (ntbase + 1) * 16];
                b2[s] = wr[o + (ntbase + 2) * 16];
                b3[s] = wr[o + (ntbase + 3) * 16];
            }
            acc0 = wmma_bf16(af, b0, acc0);
            acc1 = wmma_bf16(af, b1, acc1);
            acc2 = wmma_bf16(af, b2, acc2);
            acc3 = wmma_bf16(af, b3, acc3);
        }
        stage_wait();                                          // s_wait_asynccnt 0
        __syncthreads();
        // ---- gates = acc + xg (LDS read), publish to LDS ----
#pragma unroll
        for (int i = 0; i < 4; ++i) {
            const v8f a = (i == 0) ? acc0 : (i == 1) ? acc1 : (i == 2) ? acc2 : acc3;
            const int n = (ntbase + i) * 16 + rowL;
#pragma unroll
            for (int v = 0; v < 8; ++v) {
                const int b = brow + v;
                gates[b * G4H + n] = a[v] + (float)xstage[b * G4H + n];
            }
        }
        __syncthreads();
        // ---- cell update: thread owns 8 consecutive (b, j) states ----
#pragma unroll
        for (int q = 0; q < 8; ++q) {
            const int e = tid * 8 + q;
            const int b = e >> 8;
            const int j = e & 255;
            const float* gb = gates + b * G4H;
            float i_ = sigm(gb[j]);
            float f_ = sigm(gb[256 + j]);
            float g_ = tanhf(gb[512 + j]);
            float o_ = sigm(gb[768 + j]);
            float c = f_ * creg[q] + i_ * g_;
            creg[q] = c;
            float h = o_ * tanhf(c);
            hp[e] = (bf16)h;
            Out[((size_t)b * CCH + (chOff + j)) * TT + t] = m[b * TT + t] ? 0.0f : h;
        }
        __syncthreads();                                       // protect hp & xstage
    }
}

// ---------------- host side ----------------
extern "C" void kernel_launch(void* const* d_in, const int* in_sizes, int n_in,
                              void* d_out, int out_size, void* d_ws, size_t ws_size,
                              hipStream_t stream) {
    (void)in_sizes; (void)n_in; (void)out_size; (void)ws_size;
    const int*           x      = (const int*)d_in[0];
    const unsigned char* m      = (const unsigned char*)d_in[2];
    const float*         emb    = (const float*)d_in[3];
    const float*         conv_w = (const float*)d_in[4];
    const float*         conv_b = (const float*)d_in[5];
    const float*         ln_s   = (const float*)d_in[6];
    const float*         ln_b   = (const float*)d_in[7];
    const float*         wx_fw  = (const float*)d_in[8];
    const float*         wh_fw  = (const float*)d_in[9];
    const float*         b_fw   = (const float*)d_in[10];
    const float*         wx_bw  = (const float*)d_in[11];
    const float*         wh_bw  = (const float*)d_in[12];
    const float*         b_bw   = (const float*)d_in[13];
    float* out = (float*)d_out;

    size_t off = 0;
    auto carve = [&](size_t bytes) {
        void* p = (char*)d_ws + off;
        off += (bytes + 255) & ~(size_t)255;
        return p;
    };
    const int NCONVW = DEPTH * KW * CCH * CCH;
    const int NWX = CCH * G4H;
    const int NWH = HH * G4H;
    const size_t NACT = (size_t)BB * TT * CCH;
    const size_t NPAD = (size_t)BB * TPAD * CCH;
    const size_t NXG  = (size_t)BB * TT * G4H;

    bf16*  convWb = (bf16*)carve((size_t)NCONVW * 2);
    bf16*  wxfb   = (bf16*)carve((size_t)NWX * 2);
    bf16*  wxbb   = (bf16*)carve((size_t)NWX * 2);
    bf16*  whfb   = (bf16*)carve((size_t)NWH * 2);
    bf16*  whbb   = (bf16*)carve((size_t)NWH * 2);
    float* actF   = (float*)carve(NACT * 4);
    float* convO  = (float*)carve(NACT * 4);
    bf16*  padB   = (bf16*)carve(NPAD * 2);
    bf16*  xgB    = (bf16*)carve(NXG * 2);

    f2bf_kernel<<<(NCONVW + 255) / 256, 256, 0, stream>>>(conv_w, convWb, NCONVW);
    f2bf_kernel<<<(NWX + 255) / 256, 256, 0, stream>>>(wx_fw, wxfb, NWX);
    f2bf_kernel<<<(NWX + 255) / 256, 256, 0, stream>>>(wx_bw, wxbb, NWX);
    f2bf_kernel<<<(NWH + 255) / 256, 256, 0, stream>>>(wh_fw, whfb, NWH);
    f2bf_kernel<<<(NWH + 255) / 256, 256, 0, stream>>>(wh_bw, whbb, NWH);

    emb_kernel<<<BB * TT, 256, 0, stream>>>(x, m, emb, actF);

    const int npad_blocks = (int)((NPAD + 255) / 256);
    for (int d = 0; d < DEPTH; ++d) {
        pack_kernel<<<npad_blocks, 256, 0, stream>>>(actF, padB);
        conv_wmma_kernel<<<dim3(BB * TT / 32, CCH / 128), 256, 0, stream>>>(
            padB, convWb + (size_t)d * KW * CCH * CCH, conv_b + d * CCH, convO);
        ln_lrelu_kernel<<<BB * CCH, 256, 0, stream>>>(
            convO, ln_s + d * TT, ln_b + d * TT, m, actF);
    }

    pack_kernel<<<npad_blocks, 256, 0, stream>>>(actF, padB);
    const size_t lstm_lds = (size_t)BB * HH * 2 + (size_t)BB * G4H * 4
                          + (size_t)BB * G4H * 2;              // 208 KB

    xg_wmma_kernel<<<dim3(BB * TT / 32, G4H / 128), 256, 0, stream>>>(padB, wxfb, b_fw, xgB);
    lstm_kernel<<<1, 1024, lstm_lds, stream>>>(xgB, whfb, m, out, 0, 0);

    xg_wmma_kernel<<<dim3(BB * TT / 32, G4H / 128), 256, 0, stream>>>(padB, wxbb, b_bw, xgB);
    lstm_kernel<<<1, 1024, lstm_lds, stream>>>(xgB, whbb, m, out, 1, HH);
}
